// GATEncoder_70489003262551
// MI455X (gfx1250) — matrix operand order repaired and checked by reading, not compiled, not run on hardware
//
#include <hip/hip_runtime.h>
#include <hip/hip_bf16.h>
#include <math.h>

typedef __attribute__((ext_vector_type(16))) _Float16 v16h;
typedef __attribute__((ext_vector_type(8)))  _Float16 v8h;
typedef __attribute__((ext_vector_type(8)))  float    v8f;

#define NEG_SLOPE 0.2f

__device__ __forceinline__ float lrelu(float v) { return v > 0.f ? v : NEG_SLOPE * v; }

// ---------------- CSR build ----------------
__global__ void k_zero(int* deg, int* cur, int n) {
  int i = blockIdx.x * blockDim.x + threadIdx.x;
  if (i < n) { deg[i] = 0; cur[i] = 0; }
}

__global__ void k_degree(const int* __restrict__ ei, int* deg, int e, int n) {
  int i = blockIdx.x * blockDim.x + threadIdx.x;
  if (i >= e + n) return;
  int d = (i < e) ? ei[e + i] : (i - e);   // self-loop dst = node id
  atomicAdd(&deg[d], 1);
}

// Single-block exclusive scan over deg[0..n) -> row[0..n]
__global__ void k_scan(const int* __restrict__ deg, int* __restrict__ row, int n) {
  __shared__ int sums[1024];
  int t = threadIdx.x;
  int chunk = (n + 1023) >> 10;
  int begin = t * chunk;
  int finish = begin + chunk; if (finish > n) finish = n;
  int s = 0;
  for (int i = begin; i < finish; ++i) s += deg[i];
  sums[t] = s;
  __syncthreads();
  for (int off = 1; off < 1024; off <<= 1) {
    int v = (t >= off) ? sums[t - off] : 0;
    __syncthreads();
    sums[t] += v;
    __syncthreads();
  }
  int run = (t == 0) ? 0 : sums[t - 1];
  for (int i = begin; i < finish; ++i) { row[i] = run; run += deg[i]; }
  if (t == 1023) row[n] = sums[1023];
}

__global__ void k_fill(const int* __restrict__ ei, const int* __restrict__ row,
                       int* cur, int* __restrict__ csrc, int e, int n) {
  int i = blockIdx.x * blockDim.x + threadIdx.x;
  if (i >= e + n) return;
  int s = (i < e) ? ei[i]     : (i - e);
  int d = (i < e) ? ei[e + i] : (i - e);
  int pos = atomicAdd(&cur[d], 1);
  csrc[row[d] + pos] = s;
}

// ---------------- Layer 1 projection: h1 = x @ W1; attention logits ----------------
__global__ void __launch_bounds__(256) k_proj1(
    const float* __restrict__ x, const float* __restrict__ W1,
    const float* __restrict__ asw, const float* __restrict__ adw,
    float* __restrict__ h1, float* __restrict__ as1, float* __restrict__ ad1, int n) {
  __shared__ float sW[9 * 64];
  __shared__ float sAs[64], sAd[64];
  for (int i = threadIdx.x; i < 9 * 64; i += blockDim.x) sW[i] = W1[i];
  for (int i = threadIdx.x; i < 64; i += blockDim.x) { sAs[i] = asw[i]; sAd[i] = adw[i]; }
  __syncthreads();
  int node = blockIdx.x * blockDim.x + threadIdx.x;
  if (node >= n) return;
  float xv[9];
  #pragma unroll
  for (int k = 0; k < 9; ++k) xv[k] = x[node * 9 + k];
  float sa[4] = {0, 0, 0, 0}, da[4] = {0, 0, 0, 0};
  #pragma unroll 8
  for (int j = 0; j < 64; ++j) {
    float acc = 0.f;
    #pragma unroll
    for (int k = 0; k < 9; ++k) acc = fmaf(xv[k], sW[k * 64 + j], acc);
    h1[(size_t)node * 64 + j] = acc;
    int hh = j >> 4;
    sa[hh] = fmaf(acc, sAs[j], sa[hh]);
    da[hh] = fmaf(acc, sAd[j], da[hh]);
  }
  #pragma unroll
  for (int hh = 0; hh < 4; ++hh) { as1[node * 4 + hh] = sa[hh]; ad1[node * 4 + hh] = da[hh]; }
}

// ---------------- Layer 1 pull-aggregation + bias + ELU (4 heads x 16 ch) ----------------
__global__ void __launch_bounds__(256) k_agg1(
    const int* __restrict__ row, const int* __restrict__ csrc,
    const float* __restrict__ h1, const float* __restrict__ as1,
    const float* __restrict__ ad1, const float* __restrict__ b1,
    float* __restrict__ hact, _Float16* __restrict__ hact16, int n) {
  int lane = threadIdx.x & 31;
  int node = blockIdx.x * (blockDim.x >> 5) + (threadIdx.x >> 5);
  if (node >= n) return;
  int start = row[node], end = row[node + 1];
  float adv[4];
  #pragma unroll
  for (int h = 0; h < 4; ++h) adv[h] = ad1[node * 4 + h];
  // Phase A: per-head max over incoming edges (lane-strided + shuffle reduce)
  float m[4] = {-3.4e38f, -3.4e38f, -3.4e38f, -3.4e38f};
  for (int k = start + lane; k < end; k += 32) {
    int s = csrc[k];
    #pragma unroll
    for (int h = 0; h < 4; ++h)
      m[h] = fmaxf(m[h], lrelu(as1[s * 4 + h] + adv[h]));
  }
  #pragma unroll
  for (int off = 16; off >= 1; off >>= 1)
    #pragma unroll
    for (int h = 0; h < 4; ++h) m[h] = fmaxf(m[h], __shfl_xor(m[h], off, 32));
  // Phase B: denom = sum exp(e - m)
  float den[4] = {0, 0, 0, 0};
  for (int k = start + lane; k < end; k += 32) {
    int s = csrc[k];
    #pragma unroll
    for (int h = 0; h < 4; ++h)
      den[h] += __expf(lrelu(as1[s * 4 + h] + adv[h]) - m[h]);
  }
  #pragma unroll
  for (int off = 16; off >= 1; off >>= 1)
    #pragma unroll
    for (int h = 0; h < 4; ++h) den[h] += __shfl_xor(den[h], off, 32);
  float inv[4];
  #pragma unroll
  for (int h = 0; h < 4; ++h) inv[h] = 1.f / (den[h] + 1e-16f);
  // Phase C: weighted gather-accumulate; lane owns channels c0=lane, c1=lane+32
  int c0 = lane, c1 = lane + 32;
  int h0 = c0 >> 4, h1i = c1 >> 4;
  float acc0 = 0.f, acc1 = 0.f;
  for (int k = start; k < end; ++k) {
    int s = csrc[k];
    float a0 = __expf(lrelu(as1[s * 4 + h0] + adv[h0]) - m[h0]) * inv[h0];
    float a1 = __expf(lrelu(as1[s * 4 + h1i] + adv[h1i]) - m[h1i]) * inv[h1i];
    acc0 = fmaf(a0, h1[(size_t)s * 64 + c0], acc0);
    acc1 = fmaf(a1, h1[(size_t)s * 64 + c1], acc1);
  }
  float v0 = acc0 + b1[c0], v1 = acc1 + b1[c1];
  v0 = v0 > 0.f ? v0 : __expf(v0) - 1.f;   // ELU
  v1 = v1 > 0.f ? v1 : __expf(v1) - 1.f;
  hact[(size_t)node * 64 + c0] = v0;
  hact[(size_t)node * 64 + c1] = v1;
  hact16[(size_t)node * 64 + c0] = (_Float16)v0;
  hact16[(size_t)node * 64 + c1] = (_Float16)v1;
}

// ---------------- Layer 2 projection via WMMA f16: h2 = hact @ W2 ----------------
// One wave per 16-node tile; W2 kept transposed in LDS as f16 so B-fragments are
// contiguous ds_load_b128s; A-fragments are two 16B global loads per lane.
__global__ void __launch_bounds__(256) k_proj2(
    const _Float16* __restrict__ hact16, const float* __restrict__ W2,
    float* __restrict__ h2, int ntiles) {
  __shared__ __align__(16) _Float16 w2t[64 * 64];   // [n][k] (transposed)
  for (int i = threadIdx.x; i < 4096; i += blockDim.x) {
    int k = i >> 6, nn = i & 63;
    w2t[nn * 64 + k] = (_Float16)W2[i];
  }
  __syncthreads();
  int lane = threadIdx.x & 31;
  int tile = blockIdx.x * (blockDim.x >> 5) + (threadIdx.x >> 5);
  if (tile >= ntiles) return;                  // wave-uniform: EXEC stays all-ones
  int mrow = lane & 15, half = lane >> 4, ncol = lane & 15;
  // A fragments: 16x32 f16 layout = per lane two contiguous 8-halfword runs
  v16h afrag[2];
  const _Float16* arow = hact16 + (size_t)(tile * 16 + mrow) * 64;
  #pragma unroll
  for (int kk = 0; kk < 2; ++kk) {
    v8h lo = *(const v8h*)(arow + kk * 32 + 8 * half);
    v8h hi = *(const v8h*)(arow + kk * 32 + 16 + 8 * half);
    v16h a;
    #pragma unroll
    for (int i = 0; i < 8; ++i) { a[i] = lo[i]; a[8 + i] = hi[i]; }
    afrag[kk] = a;
  }
  #pragma unroll
  for (int nt = 0; nt < 4; ++nt) {
    v8f c = {};
    #pragma unroll
    for (int kk = 0; kk < 2; ++kk) {
      const _Float16* bp = &w2t[(nt * 16 + ncol) * 64 + kk * 32 + 8 * half];
      v8h lo = *(const v8h*)(bp);
      v8h hi = *(const v8h*)(bp + 16);
      v16h b;
      #pragma unroll
      for (int i = 0; i < 8; ++i) { b[i] = lo[i]; b[8 + i] = hi[i]; }
      c = __builtin_amdgcn_wmma_f32_16x16x32_f16(false, afrag[kk], false, b,
                                                 (short)0, c, false, false);
    }
    // D layout: VGPR r -> row r + 8*half, col = lane&15
    #pragma unroll
    for (int r = 0; r < 8; ++r)
      h2[(size_t)(tile * 16 + r + 8 * half) * 64 + nt * 16 + ncol] = c[r];
  }
}

// ---------------- Layer 2 attention logits ----------------
__global__ void k_attn2(const float* __restrict__ h2, const float* __restrict__ asw,
                        const float* __restrict__ adw, float* as2, float* ad2, int n) {
  int node = blockIdx.x * blockDim.x + threadIdx.x;
  if (node >= n) return;
  float sa = 0.f, da = 0.f;
  #pragma unroll 8
  for (int j = 0; j < 64; ++j) {
    float v = h2[(size_t)node * 64 + j];
    sa = fmaf(v, asw[j], sa);
    da = fmaf(v, adw[j], da);
  }
  as2[node] = sa; ad2[node] = da;
}

// ---------------- Layer 2 pull-aggregation (1 head x 64 ch) + bias -> out ----------------
__global__ void __launch_bounds__(256) k_agg2(
    const int* __restrict__ row, const int* __restrict__ csrc,
    const float* __restrict__ h2, const float* __restrict__ as2,
    const float* __restrict__ ad2, const float* __restrict__ b2,
    float* __restrict__ out, int n) {
  int lane = threadIdx.x & 31;
  int node = blockIdx.x * (blockDim.x >> 5) + (threadIdx.x >> 5);
  if (node >= n) return;
  int start = row[node], end = row[node + 1];
  float adv = ad2[node];
  float m = -3.4e38f;
  for (int k = start + lane; k < end; k += 32)
    m = fmaxf(m, lrelu(as2[csrc[k]] + adv));
  #pragma unroll
  for (int off = 16; off >= 1; off >>= 1) m = fmaxf(m, __shfl_xor(m, off, 32));
  float den = 0.f;
  for (int k = start + lane; k < end; k += 32)
    den += __expf(lrelu(as2[csrc[k]] + adv) - m);
  #pragma unroll
  for (int off = 16; off >= 1; off >>= 1) den += __shfl_xor(den, off, 32);
  float inv = 1.f / (den + 1e-16f);
  int c0 = lane, c1 = lane + 32;
  float acc0 = 0.f, acc1 = 0.f;
  for (int k = start; k < end; ++k) {
    int s = csrc[k];
    float a = __expf(lrelu(as2[s] + adv) - m) * inv;
    acc0 = fmaf(a, h2[(size_t)s * 64 + c0], acc0);
    acc1 = fmaf(a, h2[(size_t)s * 64 + c1], acc1);
  }
  out[(size_t)node * 64 + c0] = acc0 + b2[c0];
  out[(size_t)node * 64 + c1] = acc1 + b2[c1];
}

// ---------------- launch ----------------
static inline size_t alignup(size_t v) { return (v + 255) & ~(size_t)255; }

extern "C" void kernel_launch(void* const* d_in, const int* in_sizes, int n_in,
                              void* d_out, int out_size, void* d_ws, size_t ws_size,
                              hipStream_t stream) {
  const float* x   = (const float*)d_in[0];
  const int*   ei  = (const int*)  d_in[1];
  const float* W1  = (const float*)d_in[2];
  const float* as1w= (const float*)d_in[3];
  const float* ad1w= (const float*)d_in[4];
  const float* b1  = (const float*)d_in[5];
  const float* W2  = (const float*)d_in[6];
  const float* as2w= (const float*)d_in[7];
  const float* ad2w= (const float*)d_in[8];
  const float* b2  = (const float*)d_in[9];
  float* out = (float*)d_out;

  const int n = in_sizes[0] / 9;
  const int e = in_sizes[1] / 2;
  const int etot = e + n;

  // workspace carve-up
  char* ws = (char*)d_ws;
  size_t off = 0;
  float* h1     = (float*)(ws + off); off = alignup(off + (size_t)n * 64 * 4);
  float* hact   = (float*)(ws + off); off = alignup(off + (size_t)n * 64 * 4);
  float* h2     = (float*)(ws + off); off = alignup(off + (size_t)n * 64 * 4);
  _Float16* hact16 = (_Float16*)(ws + off); off = alignup(off + (size_t)n * 64 * 2);
  float* as1    = (float*)(ws + off); off = alignup(off + (size_t)n * 4 * 4);
  float* ad1    = (float*)(ws + off); off = alignup(off + (size_t)n * 4 * 4);
  float* as2    = (float*)(ws + off); off = alignup(off + (size_t)n * 4);
  float* ad2    = (float*)(ws + off); off = alignup(off + (size_t)n * 4);
  int* deg      = (int*)(ws + off);   off = alignup(off + (size_t)n * 4);
  int* cur      = (int*)(ws + off);   off = alignup(off + (size_t)n * 4);
  int* row      = (int*)(ws + off);   off = alignup(off + (size_t)(n + 1) * 4);
  int* csrc     = (int*)(ws + off);   off = alignup(off + (size_t)etot * 4);
  (void)ws_size; (void)n_in; (void)out_size;

  const int B = 256;
  int gN  = (n + B - 1) / B;
  int gE  = (etot + B - 1) / B;
  int waves_per_blk = B / 32;
  int gAgg = (n + waves_per_blk - 1) / waves_per_blk;
  int ntiles = (n + 15) / 16;
  int gP2 = (ntiles + waves_per_blk - 1) / waves_per_blk;

  // CSR build (int atomics only, one time)
  k_zero  <<<gN, B, 0, stream>>>(deg, cur, n);
  k_degree<<<gE, B, 0, stream>>>(ei, deg, e, n);
  k_scan  <<<1, 1024, 0, stream>>>(deg, row, n);
  k_fill  <<<gE, B, 0, stream>>>(ei, row, cur, csrc, e, n);

  // Layer 1
  k_proj1<<<gN, B, 0, stream>>>(x, W1, as1w, ad1w, h1, as1, ad1, n);
  k_agg1 <<<gAgg, B, 0, stream>>>(row, csrc, h1, as1, ad1, b1, hact, hact16, n);

  // Layer 2
  k_proj2<<<gP2, B, 0, stream>>>(hact16, W2, h2, ntiles);
  k_attn2<<<gN, B, 0, stream>>>(h2, as2w, ad2w, as2, ad2, n);
  k_agg2 <<<gAgg, B, 0, stream>>>(row, csrc, h2, as2, ad2, b2, out, n);
}
